// RelGraphConvLayer_43800076485313
// MI455X (gfx1250) — matrix operand order repaired and checked by reading, not compiled, not run on hardware
//
#include <hip/hip_runtime.h>
#include <hip/hip_bf16.h>

// ---------------------------------------------------------------------------
// RelGraphConv on MI455X (gfx1250, wave32, WMMA + async global->LDS).
// out = relu( sum_r S_r (x @ W_r) + x @ W_self + bias ),  S_r = right-norm adj.
//  1) prep: x -> bf16, W -> bf16 transposed (conversion hoisted out of GEMM)
//  2) batched WMMA GEMM, 128x64 tile, double-buffered ASYNC LDS staging
//  3) edge scatter with f32 atomics into L2-resident accumulator
//  4) bias + relu epilogue
// ---------------------------------------------------------------------------

typedef __attribute__((ext_vector_type(16))) __bf16 v16bf;
typedef __attribute__((ext_vector_type(8)))  float  v8f;

#define FEAT 256
#define NREL 4
#define BLK_M 128
#define BLK_N 64
#define TILE_K 32
#define KSTEPS (FEAT / TILE_K)
#define LDA 34   // padded bf16 row stride (keeps 16B async chunks dword-safe)

__device__ __forceinline__ unsigned short f2bf(float f) {
    unsigned int u = __float_as_uint(f);
    u += 0x7FFFu + ((u >> 16) & 1u);        // round-to-nearest-even
    return (unsigned short)(u >> 16);
}
__device__ __forceinline__ float bf2f(unsigned short h) {
    return __uint_as_float(((unsigned int)h) << 16);
}

// Async 16-byte global->LDS copy (ASYNCcnt-tracked, per CDNA5 ISA).
// LDS offset = low 32 bits of the generic pointer (flat->LDS truncates addr[31:0]).
__device__ __forceinline__ void async_copy_b128(const void* gptr, void* lptr) {
    unsigned loff = (unsigned)(size_t)lptr;
    asm volatile("global_load_async_to_lds_b128 %0, %1, off"
                 :: "v"(loff), "v"(gptr) : "memory");
}
__device__ __forceinline__ void wait_async0() {
    asm volatile("s_wait_asynccnt 0x0" ::: "memory");
}

// -------------------------------- helpers ----------------------------------
__global__ void zero_f_kernel(float* __restrict__ p, long n) {
    long i = (long)blockIdx.x * blockDim.x + threadIdx.x;
    if (i < n) p[i] = 0.0f;
}

__global__ void degree_kernel(const int* __restrict__ dst, float* __restrict__ deg,
                              int E, int N, int RE) {
    int i = blockIdx.x * blockDim.x + threadIdx.x;
    if (i < RE) {
        int r = i / E;
        atomicAdd(&deg[(size_t)r * N + dst[i]], 1.0f);
    }
}

__global__ void invdeg_kernel(float* __restrict__ deg, long n) {
    long i = (long)blockIdx.x * blockDim.x + threadIdx.x;
    if (i < n) deg[i] = 1.0f / fmaxf(deg[i], 1.0f);
}

// x (fp32) -> xbf (bf16), 4 elems/thread
__global__ void prep_x_kernel(const float4* __restrict__ x4,
                              unsigned long long* __restrict__ xbf, long n4) {
    long i = (long)blockIdx.x * blockDim.x + threadIdx.x;
    if (i < n4) {
        float4 v = x4[i];
        xbf[i] =  (unsigned long long)f2bf(v.x)
               | ((unsigned long long)f2bf(v.y) << 16)
               | ((unsigned long long)f2bf(v.z) << 32)
               | ((unsigned long long)f2bf(v.w) << 48);
    }
}

// W[r][k][n] (fp32) -> WbfT[r][n][k] (bf16); r==NREL is self-loop weight
__global__ void prep_w_kernel(const float* __restrict__ weight,
                              const float* __restrict__ selfW,
                              unsigned short* __restrict__ WbfT, int total) {
    int i = blockIdx.x * blockDim.x + threadIdx.x;   // i = ((r*256)+n)*256+k
    if (i < total) {
        int r = i >> 16;
        int n = (i >> 8) & (FEAT - 1);
        int k = i & (FEAT - 1);
        const float* W = (r < NREL) ? (weight + (size_t)r * FEAT * FEAT) : selfW;
        WbfT[i] = f2bf(W[k * FEAT + n]);
    }
}

// ------------------------- batched WMMA GEMM -------------------------------
// grid.z = r in [0, NREL]: r<NREL -> Y_r = x @ W_r (bf16), r==NREL -> acc (f32)
__global__ __launch_bounds__(256, 2)
void gemm_xw_kernel(const unsigned short* __restrict__ xbf,   // [N,256] bf16
                    const unsigned short* __restrict__ WbfT,  // [NREL+1,256,256] bf16 (n-major)
                    unsigned short* __restrict__ Ybf,         // [NREL,N,256] bf16
                    float* __restrict__ acc,                  // [N,256] f32
                    int n_rows)
{
    // double-buffered: [buf][ A(128xLDA) | BT(64xLDA) ]
    __shared__ unsigned short lds[2][BLK_M * LDA + BLK_N * LDA];

    const int r       = blockIdx.z;
    const unsigned short* Wt = WbfT + (size_t)r * FEAT * FEAT;
    const int rowBase = blockIdx.x * BLK_M;
    const int colBase = blockIdx.y * BLK_N;
    const int tid     = threadIdx.x;
    const int lane    = tid & 31;
    const int wave    = tid >> 5;           // 8 waves: 16-row subtiles
    const int hi      = (lane >> 4) & 1;
    const int l15     = lane & 15;
    const bool fullM  = (rowBase + BLK_M) <= n_rows;

    // per-thread async-copy geometry (fixed across k-steps)
    const int arow0 = tid >> 2;             // A chunk rows: tid, tid+256
    const int arow1 = (tid + 256) >> 2;
    const int asub  = (tid & 3) * 16;       // byte offset of 16B chunk in row
    const int brow  = tid >> 2;             // BT rows 0..63

    v8f cacc[4] = {};                       // 16x64 per wave

    // stage tile ks into buffer buf (async, 3 instrs/thread)
    auto stage = [&](int ks, unsigned short* buf) {
        const char* gA = (const char*)(xbf) + asub + (size_t)ks * TILE_K * 2;
        const char* gB = (const char*)(Wt + (size_t)(colBase + brow) * FEAT
                                          + ks * TILE_K) + asub;
        unsigned short* bufA = buf;
        unsigned short* bufB = buf + BLK_M * LDA;
        int g0 = rowBase + arow0, g1 = rowBase + arow1;
        if (fullM || g0 < n_rows)
            async_copy_b128(gA + (size_t)g0 * FEAT * 2, bufA + arow0 * LDA) ,
            (void)0;
        if (fullM || g1 < n_rows)
            async_copy_b128(gA + (size_t)g1 * FEAT * 2, bufA + arow1 * LDA);
        async_copy_b128(gB, bufB + brow * LDA);
    };

    // fix: asub must also offset the LDS side of A/B chunks
    // (handled below by re-deriving pointers inside stage2)
    auto stage2 = [&](int ks, unsigned short* buf) {
        const char* gxk = (const char*)xbf + (size_t)ks * TILE_K * 2 + asub;
        unsigned short* bufA = buf;
        unsigned short* bufB = buf + BLK_M * LDA;
        int g0 = rowBase + arow0, g1 = rowBase + arow1;
        if (fullM || g0 < n_rows)
            async_copy_b128(gxk + (size_t)g0 * FEAT * 2,
                            (char*)(bufA + arow0 * LDA) + asub);
        if (fullM || g1 < n_rows)
            async_copy_b128(gxk + (size_t)g1 * FEAT * 2,
                            (char*)(bufA + arow1 * LDA) + asub);
        async_copy_b128((const char*)(Wt + (size_t)(colBase + brow) * FEAT
                                         + ks * TILE_K) + asub,
                        (char*)(bufB + brow * LDA) + asub);
    };
    (void)stage;

    // prologue: stage tile 0
    stage2(0, &lds[0][0]);
    wait_async0();
    __syncthreads();

    for (int ks = 0; ks < KSTEPS; ++ks) {
        unsigned short* cbuf = &lds[ks & 1][0];
        if (ks + 1 < KSTEPS)
            stage2(ks + 1, &lds[(ks + 1) & 1][0]);   // overlap with compute

        // ---- fragments per ISA 7.12.2 VGPR layouts (2 bf16 / u32 load) ----
        union { v16bf v; unsigned int u[8]; } af;
        const unsigned short* pa = cbuf + (wave * 16 + l15) * LDA;
        #pragma unroll
        for (int v = 0; v < 8; ++v) {
            int kA = 2 * v + 8 * (hi + ((v >= 4) ? 1 : 0));  // A 16x32 table
            af.u[v] = *(const unsigned int*)(pa + kA);
        }
        #pragma unroll
        for (int nt = 0; nt < 4; ++nt) {
            union { v16bf v; unsigned int u[8]; } bfm;
            const unsigned short* pb = cbuf + BLK_M * LDA + (nt * 16 + l15) * LDA;
            #pragma unroll
            for (int v = 0; v < 8; ++v) {
                int kB = 2 * v + 8 * (hi + ((v >= 4) ? 1 : 0));  // B 32x16 table
                bfm.u[v] = *(const unsigned int*)(pb + kB);
            }
            cacc[nt] = __builtin_amdgcn_wmma_f32_16x16x32_bf16(
                           false, af.v, false, bfm.v, (short)0, cacc[nt], false, false);
        }
        wait_async0();
        __syncthreads();
    }

    // D layout: VGPR v -> M = v + 8*hi, N = lane&15
    #pragma unroll
    for (int nt = 0; nt < 4; ++nt) {
        #pragma unroll
        for (int v = 0; v < 8; ++v) {
            int row = rowBase + wave * 16 + v + 8 * hi;
            int col = colBase + nt * 16 + l15;
            if (fullM || row < n_rows) {
                if (r < NREL)
                    Ybf[((size_t)r * n_rows + row) * FEAT + col] = f2bf(cacc[nt][v]);
                else
                    acc[(size_t)row * FEAT + col] = cacc[nt][v];
            }
        }
    }
}

// ------------------------- edge scatter (L2-resident target) ---------------
// 4 edges per 256-thread block; each thread: one b64 load (4 bf16) + 4 atomics.
__global__ __launch_bounds__(256)
void scatter_kernel(const unsigned short* __restrict__ Ybf,
                    const int* __restrict__ src, const int* __restrict__ dst,
                    const float* __restrict__ invdeg, float* __restrict__ acc,
                    int E, int N, int RE)
{
    int t  = threadIdx.x;
    int be = blockIdx.x * 4 + (t >> 6);       // r*E + e
    if (be >= RE) return;
    int r = be / E;
    int s = src[be];
    int d = dst[be];
    float w = invdeg[(size_t)r * N + d];
    int f = (t & 63) * 4;
    unsigned long long pk =
        *(const unsigned long long*)(Ybf + ((size_t)r * N + s) * FEAT + f);
    float* ap = acc + (size_t)d * FEAT + f;
    atomicAdd(ap + 0, bf2f((unsigned short)(pk      )) * w);
    atomicAdd(ap + 1, bf2f((unsigned short)(pk >> 16)) * w);
    atomicAdd(ap + 2, bf2f((unsigned short)(pk >> 32)) * w);
    atomicAdd(ap + 3, bf2f((unsigned short)(pk >> 48)) * w);
}

// ------------------------- bias + relu epilogue (float4) -------------------
__global__ void bias_relu_kernel(const float4* __restrict__ acc,
                                 const float* __restrict__ bias,
                                 float4* __restrict__ out, long total4)
{
    long i = (long)blockIdx.x * blockDim.x + threadIdx.x;
    if (i < total4) {
        float4 a = acc[i];
        int f = (int)((i * 4) & (FEAT - 1));
        a.x += bias[f + 0]; a.y += bias[f + 1];
        a.z += bias[f + 2]; a.w += bias[f + 3];
        a.x = a.x > 0.f ? a.x : 0.f;
        a.y = a.y > 0.f ? a.y : 0.f;
        a.z = a.z > 0.f ? a.z : 0.f;
        a.w = a.w > 0.f ? a.w : 0.f;
        out[i] = a;
    }
}

// ---------------------------------------------------------------------------
extern "C" void kernel_launch(void* const* d_in, const int* in_sizes, int n_in,
                              void* d_out, int out_size, void* d_ws, size_t ws_size,
                              hipStream_t stream)
{
    const float* x      = (const float*)d_in[0];
    const int*   src    = (const int*)  d_in[1];
    const int*   dst    = (const int*)  d_in[2];
    const float* weight = (const float*)d_in[3];
    const float* selfW  = (const float*)d_in[4];
    const float* bias   = (const float*)d_in[5];

    const int N  = in_sizes[0] / FEAT;
    const int E  = in_sizes[1] / NREL;
    const int RE = NREL * E;

    // workspace layout
    char* ws = (char*)d_ws;
    size_t off = 0;
    unsigned short* Ybf  = (unsigned short*)(ws + off); off += (size_t)NREL * N * FEAT * 2;
    float*          acc  = (float*)         (ws + off); off += (size_t)N * FEAT * 4;
    float*          deg  = (float*)         (ws + off); off += (size_t)NREL * N * 4;
    unsigned short* xbf  = (unsigned short*)(ws + off); off += (size_t)N * FEAT * 2;
    unsigned short* WbfT = (unsigned short*)(ws + off);

    long degN = (long)NREL * N;
    zero_f_kernel<<<(int)((degN + 255) / 256), 256, 0, stream>>>(deg, degN);
    degree_kernel<<<(RE + 255) / 256, 256, 0, stream>>>(dst, deg, E, N, RE);
    invdeg_kernel<<<(int)((degN + 255) / 256), 256, 0, stream>>>(deg, degN);

    long n4 = (long)N * FEAT / 4;
    prep_x_kernel<<<(int)((n4 + 255) / 256), 256, 0, stream>>>(
        (const float4*)x, (unsigned long long*)xbf, n4);
    int wtotal = (NREL + 1) * FEAT * FEAT;
    prep_w_kernel<<<(wtotal + 255) / 256, 256, 0, stream>>>(weight, selfW, WbfT, wtotal);

    dim3 grid((N + BLK_M - 1) / BLK_M, FEAT / BLK_N, NREL + 1);
    gemm_xw_kernel<<<grid, 256, 0, stream>>>(xbf, WbfT, Ybf, acc, N);

    scatter_kernel<<<(RE + 3) / 4, 256, 0, stream>>>(Ybf, src, dst, deg, acc, E, N, RE);

    long total4 = (long)N * FEAT / 4;
    bias_relu_kernel<<<(int)((total4 + 255) / 256), 256, 0, stream>>>(
        (const float4*)acc, bias, (float4*)d_out, total4);
}
